// researchModel_12283606468193
// MI455X (gfx1250) — matrix operand order repaired and checked by reading, not compiled, not run on hardware
//
#include <hip/hip_runtime.h>
#include <hip/hip_bf16.h>

typedef __bf16 bf16;
typedef __attribute__((ext_vector_type(16))) __bf16 v16bf;
typedef __attribute__((ext_vector_type(8)))  __bf16 v8bf;
typedef __attribute__((ext_vector_type(8)))  float  v8f;
typedef __attribute__((vector_size(16)))     int    v4i;

#define NN      8192
#define FH      512      // h row width (F1+F2)
#define OC      256      // concat output cols (O1+O2), also MLP out
#define HEADC   128      // cols per head
#define ALPHA   0.2f
#define MLP_SLOPE 0.01f
#define NEG_INF (-9e15f)

#define KTILE   128              // K per attn iteration (4 WMMAs)
#define NIT     (NN / KTILE)     // 64

#define AS1 __attribute__((address_space(1)))
#define AS3 __attribute__((address_space(3)))

union V16U { v16bf v; v8bf h[2]; };

static __device__ __forceinline__ float lrelu(float x, float s) {
    return x >= 0.0f ? x : s * x;
}

// ---- CDNA5 async copy: global -> LDS (16B per lane), ASYNCcnt-tracked ----
static __device__ __forceinline__ void async_cp16(const int* g, int* l) {
#if __has_builtin(__builtin_amdgcn_global_load_async_to_lds_b128)
    __builtin_amdgcn_global_load_async_to_lds_b128((AS1 v4i*)g, (AS3 v4i*)l, 0, 0);
#else
    unsigned lofs = (unsigned)(size_t)(AS3 void*)l;
    asm volatile("global_load_async_to_lds_b128 %0, %1, off"
                 :: "v"(lofs), "v"(g) : "memory");
#endif
}

#if __has_builtin(__builtin_amdgcn_s_wait_asynccnt)
#define WAIT_ASYNC(n) __builtin_amdgcn_s_wait_asynccnt(n)
#else
#define WAIT_ASYNC(n) asm volatile("s_wait_asynccnt " #n ::: "memory")
#endif

// ---------------------------------------------------------------------------
// Kernel 1: wa vectors.  wa1s[k] = sum_c W1[k][c]*a1[c], etc.  (1 block, 256 thr)
// wa layout: [0:256)=wa1_src [256:512)=wa1_dst [512:768)=wa2_src [768:1024)=wa2_dst
__global__ void prep_wa_kernel(const float* __restrict__ W1, const float* __restrict__ W2,
                               const float* __restrict__ a1, const float* __restrict__ a2,
                               float* __restrict__ wa) {
    int k = threadIdx.x;                       // 0..255
    float p0 = 0.f, p1 = 0.f, p2 = 0.f, p3 = 0.f;
    for (int c = 0; c < HEADC; ++c) {
        float w1 = W1[k * HEADC + c];
        float w2 = W2[k * HEADC + c];
        p0 += w1 * a1[c];
        p1 += w1 * a1[HEADC + c];
        p2 += w2 * a2[c];
        p3 += w2 * a2[HEADC + c];
    }
    wa[k] = p0; wa[256 + k] = p1; wa[512 + k] = p2; wa[768 + k] = p3;
}

// ---------------------------------------------------------------------------
// Kernel 2: Wh = x_head @ W_head, written TRANSPOSED as bf16: whT[col(0..255)][row(0..8191)]
// grid (512, 2 heads), block 256 (8 waves).  WMMA bf16 16x16x32.
__global__ void wh_kernel(const float* __restrict__ h, const float* __restrict__ W1,
                          const float* __restrict__ W2, bf16* __restrict__ whT) {
    const int head = blockIdx.y;
    const int i0   = blockIdx.x * 16;
    const int wave = threadIdx.x >> 5;
    const int lane = threadIdx.x & 31;
    const int laneHi = lane >> 4;
    const int nloc   = wave * 16 + (lane & 15);   // 0..127 within head
    const float* W   = head == 0 ? W1 : W2;
    const int mrow   = lane & 15;

    v8f c = {0.f, 0.f, 0.f, 0.f, 0.f, 0.f, 0.f, 0.f};

    for (int kb = 0; kb < 8; ++kb) {              // K = 256, step 32
        // ---- A fragment: h[i0+mrow][head*256 + kb*32 + K(e)]
        const float* abase = h + (size_t)(i0 + mrow) * FH + head * 256 + kb * 32;
        float4 g0 = *(const float4*)(abase + laneHi * 8);
        float4 g1 = *(const float4*)(abase + laneHi * 8 + 4);
        float4 g2 = *(const float4*)(abase + 16 + laneHi * 8);
        float4 g3 = *(const float4*)(abase + 16 + laneHi * 8 + 4);
        v16bf a;
        a[0]=(bf16)g0.x; a[1]=(bf16)g0.y; a[2]=(bf16)g0.z; a[3]=(bf16)g0.w;
        a[4]=(bf16)g1.x; a[5]=(bf16)g1.y; a[6]=(bf16)g1.z; a[7]=(bf16)g1.w;
        a[8]=(bf16)g2.x; a[9]=(bf16)g2.y; a[10]=(bf16)g2.z; a[11]=(bf16)g2.w;
        a[12]=(bf16)g3.x; a[13]=(bf16)g3.y; a[14]=(bf16)g3.z; a[15]=(bf16)g3.w;
        // ---- B fragment: W[kb*32 + K(e)][nloc], K(e) = laneHi*8 + e (+8 for e>=8)
        v16bf b;
        #pragma unroll
        for (int e = 0; e < 16; ++e)
            b[e] = (bf16)W[(size_t)(kb * 32 + laneHi * 8 + e + (e >= 8 ? 8 : 0)) * HEADC + nloc];
        c = __builtin_amdgcn_wmma_f32_16x16x32_bf16(false, a, false, b,
                                                    (short)0, c, false, false);
    }
    // ---- store C transposed: whT[head*128+nloc][i0 + 8*laneHi + v], v=0..7 contiguous
    v8bf outp;
    #pragma unroll
    for (int v = 0; v < 8; ++v) outp[v] = (bf16)c[v];
    bf16* dst = whT + (size_t)(head * HEADC + nloc) * NN + i0 + 8 * laneHi;
    *(v8bf*)dst = outp;
}

// ---------------------------------------------------------------------------
// Kernel 3: per-row scores.  1 row per wave; grid 1024, block 256.
__global__ void scores_kernel(const float* __restrict__ h, const float* __restrict__ wa,
                              float* __restrict__ s1s, float* __restrict__ s1d,
                              float* __restrict__ s2s, float* __restrict__ s2d) {
    const int wave = threadIdx.x >> 5;
    const int lane = threadIdx.x & 31;
    const int row  = blockIdx.x * 8 + wave;
    const float* wa1s = wa, *wa1d = wa + 256, *wa2s = wa + 512, *wa2d = wa + 768;
    float p0 = 0.f, p1 = 0.f, p2 = 0.f, p3 = 0.f;
    const float* hr = h + (size_t)row * FH;
    for (int k = lane; k < 256; k += 32) {
        float x1 = hr[k], x2 = hr[256 + k];
        p0 += x1 * wa1s[k]; p1 += x1 * wa1d[k];
        p2 += x2 * wa2s[k]; p3 += x2 * wa2d[k];
    }
    #pragma unroll
    for (int off = 16; off; off >>= 1) {
        p0 += __shfl_xor(p0, off);
        p1 += __shfl_xor(p1, off);
        p2 += __shfl_xor(p2, off);
        p3 += __shfl_xor(p3, off);
    }
    if (lane == 0) { s1s[row] = p0; s1d[row] = p1; s2s[row] = p2; s2d[row] = p3; }
}

// ---------------------------------------------------------------------------
// Kernel 4: softmax stats (row max + inverse sum) for both heads.
// grid 8192 (one block per row), block 256.  adj row mask cached in LDS, int4 loads.
__global__ void stats_kernel(const int* __restrict__ adj,
                             const float* __restrict__ s1s, const float* __restrict__ s1d,
                             const float* __restrict__ s2s, const float* __restrict__ s2d,
                             float* __restrict__ m1, float* __restrict__ z1,
                             float* __restrict__ m2, float* __restrict__ z2) {
    __shared__ unsigned char msk[NN];
    __shared__ float2 red[256];
    const int row = blockIdx.x, t = threadIdx.x;
    const float a1 = s1s[row], a2 = s2s[row];
    const int* adjrow = adj + (size_t)row * NN;
    float mx1 = NEG_INF, mx2 = NEG_INF;
    for (int j = t * 4; j < NN; j += 1024) {
        int4   av = *(const int4*)(adjrow + j);
        float4 d1 = *(const float4*)(s1d + j);
        float4 d2 = *(const float4*)(s2d + j);
        uchar4 mk;
        mk.x = av.x > 0; mk.y = av.y > 0; mk.z = av.z > 0; mk.w = av.w > 0;
        *(uchar4*)&msk[j] = mk;
        float v;
        v = mk.x ? lrelu(a1 + d1.x, ALPHA) : NEG_INF; mx1 = fmaxf(mx1, v);
        v = mk.y ? lrelu(a1 + d1.y, ALPHA) : NEG_INF; mx1 = fmaxf(mx1, v);
        v = mk.z ? lrelu(a1 + d1.z, ALPHA) : NEG_INF; mx1 = fmaxf(mx1, v);
        v = mk.w ? lrelu(a1 + d1.w, ALPHA) : NEG_INF; mx1 = fmaxf(mx1, v);
        v = mk.x ? lrelu(a2 + d2.x, ALPHA) : NEG_INF; mx2 = fmaxf(mx2, v);
        v = mk.y ? lrelu(a2 + d2.y, ALPHA) : NEG_INF; mx2 = fmaxf(mx2, v);
        v = mk.z ? lrelu(a2 + d2.z, ALPHA) : NEG_INF; mx2 = fmaxf(mx2, v);
        v = mk.w ? lrelu(a2 + d2.w, ALPHA) : NEG_INF; mx2 = fmaxf(mx2, v);
    }
    red[t] = make_float2(mx1, mx2);
    __syncthreads();
    for (int s = 128; s > 0; s >>= 1) {
        if (t < s) {
            red[t].x = fmaxf(red[t].x, red[t + s].x);
            red[t].y = fmaxf(red[t].y, red[t + s].y);
        }
        __syncthreads();
    }
    const float M1 = red[0].x, M2 = red[0].y;
    __syncthreads();
    float sm1 = 0.f, sm2 = 0.f;
    for (int j = t * 4; j < NN; j += 1024) {
        uchar4 mk = *(const uchar4*)&msk[j];
        float4 d1 = *(const float4*)(s1d + j);
        float4 d2 = *(const float4*)(s2d + j);
        sm1 += __expf((mk.x ? lrelu(a1 + d1.x, ALPHA) : NEG_INF) - M1);
        sm1 += __expf((mk.y ? lrelu(a1 + d1.y, ALPHA) : NEG_INF) - M1);
        sm1 += __expf((mk.z ? lrelu(a1 + d1.z, ALPHA) : NEG_INF) - M1);
        sm1 += __expf((mk.w ? lrelu(a1 + d1.w, ALPHA) : NEG_INF) - M1);
        sm2 += __expf((mk.x ? lrelu(a2 + d2.x, ALPHA) : NEG_INF) - M2);
        sm2 += __expf((mk.y ? lrelu(a2 + d2.y, ALPHA) : NEG_INF) - M2);
        sm2 += __expf((mk.z ? lrelu(a2 + d2.z, ALPHA) : NEG_INF) - M2);
        sm2 += __expf((mk.w ? lrelu(a2 + d2.w, ALPHA) : NEG_INF) - M2);
    }
    red[t] = make_float2(sm1, sm2);
    __syncthreads();
    for (int s = 128; s > 0; s >>= 1) {
        if (t < s) {
            red[t].x += red[t + s].x;
            red[t].y += red[t + s].y;
        }
        __syncthreads();
    }
    if (t == 0) {
        m1[row] = M1; z1[row] = 1.0f / red[0].x;
        m2[row] = M2; z2[row] = 1.0f / red[0].y;
    }
}

// ---------------------------------------------------------------------------
// Kernel 5: out_head = softmax(e) @ Wh.  adj tiles async-prefetched to LDS
// (double-buffered); 16x128 att tile built per iteration; 4 WMMAs per wave.
// grid (512 row tiles, 2 heads), block 256 (8 waves; wave w -> cols 16w..16w+15).
__global__ void attn_kernel(const int* __restrict__ adj, const bf16* __restrict__ whT,
                            const float* __restrict__ s1s, const float* __restrict__ s1d,
                            const float* __restrict__ s2s, const float* __restrict__ s2d,
                            const float* __restrict__ m1, const float* __restrict__ z1,
                            const float* __restrict__ m2, const float* __restrict__ z2,
                            bf16* __restrict__ hpr) {
    __shared__ __align__(16) int  adjbuf[2][16 * KTILE];   // 2 x 8KB
    __shared__ __align__(16) bf16 A_lds[16][KTILE];        // 4KB
    const int head = blockIdx.y;
    const int i0   = blockIdx.x * 16;
    const int t    = threadIdx.x;
    const int wave = t >> 5, lane = t & 31, laneHi = lane >> 4;
    const int mrow = lane & 15;
    const float* ssrc = head ? s2s : s1s;
    const float* sdst = head ? s2d : s1d;
    const float* m    = head ? m2  : m1;
    const float* zi   = head ? z2  : z1;

    // builder mapping: thread t owns row brow = t>>4, k-slice bk..bk+7
    const int brow = t >> 4;
    const int bk   = (t & 15) * 8;
    const float srcR = ssrc[i0 + brow];
    const float mR   = m[i0 + brow];
    const float zR   = zi[i0 + brow];
    const int* adjrow = adj + (size_t)(i0 + brow) * NN;
    int* lslot = &adjbuf[0][brow * KTILE + bk];

    // prologue: prefetch tile 0
    async_cp16(adjrow + bk,     lslot);
    async_cp16(adjrow + bk + 4, lslot + 4);

    v8f c = {0.f, 0.f, 0.f, 0.f, 0.f, 0.f, 0.f, 0.f};
    const int colg = head * HEADC + wave * 16 + mrow;
    const bf16* bbase = whT + (size_t)colg * NN;

    for (int kb = 0; kb < NIT; ++kb) {
        const int k0  = kb * KTILE;
        const int cur = kb & 1;
        if (kb + 1 < NIT) {
            int* nslot = &adjbuf[cur ^ 1][brow * KTILE + bk];
            async_cp16(adjrow + k0 + KTILE + bk,     nslot);
            async_cp16(adjrow + k0 + KTILE + bk + 4, nslot + 4);
            WAIT_ASYNC(2);               // tile kb complete (in-order)
        } else {
            WAIT_ASYNC(0);
        }
        __syncthreads();                 // adj tile ready; prev A reads done
        // ---- build 8 att weights for (brow, bk..bk+7) ----
        float4 sd0 = *(const float4*)(sdst + k0 + bk);
        float4 sd1 = *(const float4*)(sdst + k0 + bk + 4);
        int4 av0 = *(const int4*)&adjbuf[cur][brow * KTILE + bk];
        int4 av1 = *(const int4*)&adjbuf[cur][brow * KTILE + bk + 4];
        v8bf wrow;
        {
            float v0 = av0.x > 0 ? lrelu(srcR + sd0.x, ALPHA) : NEG_INF;
            float v1 = av0.y > 0 ? lrelu(srcR + sd0.y, ALPHA) : NEG_INF;
            float v2 = av0.z > 0 ? lrelu(srcR + sd0.z, ALPHA) : NEG_INF;
            float v3 = av0.w > 0 ? lrelu(srcR + sd0.w, ALPHA) : NEG_INF;
            float v4 = av1.x > 0 ? lrelu(srcR + sd1.x, ALPHA) : NEG_INF;
            float v5 = av1.y > 0 ? lrelu(srcR + sd1.y, ALPHA) : NEG_INF;
            float v6 = av1.z > 0 ? lrelu(srcR + sd1.z, ALPHA) : NEG_INF;
            float v7 = av1.w > 0 ? lrelu(srcR + sd1.w, ALPHA) : NEG_INF;
            wrow[0] = (bf16)(__expf(v0 - mR) * zR);
            wrow[1] = (bf16)(__expf(v1 - mR) * zR);
            wrow[2] = (bf16)(__expf(v2 - mR) * zR);
            wrow[3] = (bf16)(__expf(v3 - mR) * zR);
            wrow[4] = (bf16)(__expf(v4 - mR) * zR);
            wrow[5] = (bf16)(__expf(v5 - mR) * zR);
            wrow[6] = (bf16)(__expf(v6 - mR) * zR);
            wrow[7] = (bf16)(__expf(v7 - mR) * zR);
        }
        *(v8bf*)&A_lds[brow][bk] = wrow;
        __syncthreads();                 // A tile ready
        // ---- 4 back-to-back WMMAs ----
        #pragma unroll
        for (int j = 0; j < 4; ++j) {
            const v8bf* rowp = (const v8bf*)&A_lds[mrow][j * 32];
            V16U af;
            af.h[0] = rowp[2 * laneHi];        // K = j*32 + laneHi*16 .. +7
            af.h[1] = rowp[2 * laneHi + 1];    // K = j*32 + laneHi*16 + 8 .. +15
            v16bf b = *(const v16bf*)(bbase + k0 + j * 32 + laneHi * 16);
            c = __builtin_amdgcn_wmma_f32_16x16x32_bf16(false, af.v, false, b,
                                                        (short)0, c, false, false);
        }
    }
    // ---- epilogue: ELU, bf16 store to hpr[row][colg] ----
    #pragma unroll
    for (int v = 0; v < 8; ++v) {
        int i = i0 + 8 * laneHi + v;
        float x = c[v];
        x = x > 0.0f ? x : expm1f(x);
        hpr[(size_t)i * OC + colg] = (bf16)x;
    }
}

// ---------------------------------------------------------------------------
// Kernel 6: out = leaky(hpr @ mlp_W + b, 0.01).  grid 512, block 512 (16 waves = 256 cols).
__global__ void mlp_kernel(const bf16* __restrict__ hpr, const float* __restrict__ mlp_W,
                           const float* __restrict__ mlp_b, float* __restrict__ out) {
    const int i0   = blockIdx.x * 16;
    const int wave = threadIdx.x >> 5;
    const int lane = threadIdx.x & 31;
    const int laneHi = lane >> 4;
    const int colg   = wave * 16 + (lane & 15);
    const int mrow   = lane & 15;

    v8f c = {0.f, 0.f, 0.f, 0.f, 0.f, 0.f, 0.f, 0.f};
    for (int kb = 0; kb < 8; ++kb) {          // K = 256, step 32
        const bf16* abase = hpr + (size_t)(i0 + mrow) * OC + kb * 32;
        V16U af;
        af.h[0] = *(const v8bf*)(abase + laneHi * 16);
        af.h[1] = *(const v8bf*)(abase + laneHi * 16 + 8);
        v16bf b;
        #pragma unroll
        for (int e = 0; e < 16; ++e)
            b[e] = (bf16)mlp_W[(size_t)(kb * 32 + laneHi * 16 + e) * OC + colg];
        c = __builtin_amdgcn_wmma_f32_16x16x32_bf16(false, af.v, false, b,
                                                    (short)0, c, false, false);
    }
    const float bias = mlp_b[colg];
    #pragma unroll
    for (int v = 0; v < 8; ++v) {
        int i = i0 + 8 * laneHi + v;
        float x = c[v] + bias;
        out[(size_t)i * OC + colg] = lrelu(x, MLP_SLOPE);
    }
}

// ---------------------------------------------------------------------------
extern "C" void kernel_launch(void* const* d_in, const int* in_sizes, int n_in,
                              void* d_out, int out_size, void* d_ws, size_t ws_size,
                              hipStream_t stream) {
    const float* h     = (const float*)d_in[0];
    const int*   adj   = (const int*)d_in[1];
    const float* W1    = (const float*)d_in[2];
    const float* W2    = (const float*)d_in[3];
    const float* a1    = (const float*)d_in[4];
    const float* a2    = (const float*)d_in[5];
    const float* mlp_W = (const float*)d_in[6];
    const float* mlp_b = (const float*)d_in[7];
    float* out = (float*)d_out;

    char* ws = (char*)d_ws;
    size_t off = 0;
    bf16* whT = (bf16*)(ws + off); off += (size_t)OC * NN * sizeof(bf16);   // 4 MB
    bf16* hpr = (bf16*)(ws + off); off += (size_t)NN * OC * sizeof(bf16);   // 4 MB
    float* s1s = (float*)(ws + off); off += NN * sizeof(float);
    float* s1d = (float*)(ws + off); off += NN * sizeof(float);
    float* s2s = (float*)(ws + off); off += NN * sizeof(float);
    float* s2d = (float*)(ws + off); off += NN * sizeof(float);
    float* m1  = (float*)(ws + off); off += NN * sizeof(float);
    float* z1  = (float*)(ws + off); off += NN * sizeof(float);
    float* m2  = (float*)(ws + off); off += NN * sizeof(float);
    float* z2  = (float*)(ws + off); off += NN * sizeof(float);
    float* wa  = (float*)(ws + off); off += 1024 * sizeof(float);

    prep_wa_kernel<<<1, 256, 0, stream>>>(W1, W2, a1, a2, wa);
    wh_kernel<<<dim3(NN / 16, 2), 256, 0, stream>>>(h, W1, W2, whT);
    scores_kernel<<<NN / 8, 256, 0, stream>>>(h, wa, s1s, s1d, s2s, s2d);
    stats_kernel<<<NN, 256, 0, stream>>>(adj, s1s, s1d, s2s, s2d, m1, z1, m2, z2);
    attn_kernel<<<dim3(NN / 16, 2), 256, 0, stream>>>(adj, whT, s1s, s1d, s2s, s2d,
                                                      m1, z1, m2, z2, hpr);
    mlp_kernel<<<NN / 16, 512, 0, stream>>>(hpr, mlp_W, mlp_b, out);
}
